// San_44659069944117
// MI455X (gfx1250) — matrix operand order repaired and checked by reading, not compiled, not run on hardware
//
#include <hip/hip_runtime.h>

// ---------------------------------------------------------------------------
// Types for CDNA5 WMMA
// ---------------------------------------------------------------------------
typedef __bf16 bf16_t;
typedef bf16_t v16bf __attribute__((ext_vector_type(16)));
typedef bf16_t v8bf  __attribute__((ext_vector_type(8)));
typedef bf16_t v2bf  __attribute__((ext_vector_type(2)));
typedef float  v8f   __attribute__((ext_vector_type(8)));

#define NHID 4096
#define NTH  4096
#define NTP  4096

#define LDSB 40   // bf16 stride: 80 bytes -> 16B aligned rows, bank-conflict free

// ---- CDNA5 async global->LDS copy (ASYNCcnt-tracked DMA path) --------------
typedef __attribute__((address_space(3))) char lds_char;
__device__ __forceinline__ unsigned lds_addr_of(const void* p) {
    return (unsigned)(unsigned long long)(lds_char*)p;   // flat -> LDS offset
}
__device__ __forceinline__ void async_cp16(unsigned lds_dst, const void* gsrc) {
    asm volatile("global_load_async_to_lds_b128 %0, %1, off"
                 :: "v"(lds_dst), "v"((unsigned long long)gsrc) : "memory");
}
__device__ __forceinline__ void async_wait0() {
    asm volatile("s_wait_asynccnt 0x0" ::: "memory");
}

// ---------------------------------------------------------------------------
// C = W2^T @ M_p   (4096x4096x4096, f32 in/out, bf16 WMMA inner product)
// Block: 256 threads = 8 waves, 128x128 C tile.
// Pipeline per K-step of 32:
//   global_load_async_to_lds_b128 (raw f32 tiles, next K-step)
//   ds_load_b128 fragments (bf16, current) -> 8x v_wmma_f32_16x16x32_bf16
//   s_wait_asynccnt 0 ; barrier ; repack f32[k][x] -> bf16[x][k] (cvt_pk)
// ---------------------------------------------------------------------------
__global__ __launch_bounds__(256)
void k_wmma_gemm(float* __restrict__ C, const float* __restrict__ W2,
                 const float* __restrict__ Mp, int N) {
    __shared__ float  stg_a[2][32 * 128];      // raw f32 tile [k][x] (async dst)
    __shared__ float  stg_b[2][32 * 128];
    __shared__ bf16_t bfa[2][128 * LDSB];      // bf16 tile [x][k] (WMMA feed)
    __shared__ bf16_t bfb[2][128 * LDSB];

    const int tid   = threadIdx.x;
    const int lane  = tid & 31;
    const int wave  = tid >> 5;
    const int lm    = lane & 15;
    const int hi    = (lane >> 4) & 1;
    const int waveM = wave >> 1;        // 0..3
    const int waveN = wave & 1;         // 0..1
    const int i0    = blockIdx.y * 128; // C row block  (columns of W2)
    const int j0    = blockIdx.x * 128; // C col block  (columns of M_p)

    v8f acc[2][4];
    const v8f vz = {0.f, 0.f, 0.f, 0.f, 0.f, 0.f, 0.f, 0.f};
#pragma unroll
    for (int mt = 0; mt < 2; ++mt)
#pragma unroll
        for (int nt = 0; nt < 4; ++nt) acc[mt][nt] = vz;

    // ---- prologue: async-stage K-tile 0, then repack to bf16
#pragma unroll
    for (int c = 0; c < 4; ++c) {               // 1024 16B chunks per matrix
        const int q  = tid + c * 256;
        const int k  = q >> 5;                  // 32 chunks per 512B row
        const int xb = (q & 31) << 4;           // byte offset within row
        async_cp16(lds_addr_of(&stg_a[0][k * 128]) + xb,
                   (const char*)&W2[(size_t)k * N + i0] + xb);
        async_cp16(lds_addr_of(&stg_b[0][k * 128]) + xb,
                   (const char*)&Mp[(size_t)k * N + j0] + xb);
    }
    async_wait0();
    __syncthreads();
#pragma unroll
    for (int c = 0; c < 8; ++c) {               // 2048 (k-pair, x) entries
        const int p  = tid + c * 256;
        const int k2 = (p >> 7) << 1;
        const int x  = p & 127;
        const v2bf pa = {(bf16_t)stg_a[0][k2 * 128 + x], (bf16_t)stg_a[0][(k2 + 1) * 128 + x]};
        const v2bf pb = {(bf16_t)stg_b[0][k2 * 128 + x], (bf16_t)stg_b[0][(k2 + 1) * 128 + x]};
        *(v2bf*)&bfa[0][x * LDSB + k2] = pa;
        *(v2bf*)&bfb[0][x * LDSB + k2] = pb;
    }
    __syncthreads();

    int cur = 0;
    for (int kk = 0; kk < N; kk += 32) {
        const bool more = (kk + 32) < N;
        const int  nx   = cur ^ 1;

        // ---- DMA next K-tile into the alternate staging buffer (overlaps WMMA)
        if (more) {
#pragma unroll
            for (int c = 0; c < 4; ++c) {
                const int q  = tid + c * 256;
                const int k  = q >> 5;
                const int xb = (q & 31) << 4;
                async_cp16(lds_addr_of(&stg_a[nx][k * 128]) + xb,
                           (const char*)&W2[(size_t)(kk + 32 + k) * N + i0] + xb);
                async_cp16(lds_addr_of(&stg_b[nx][k * 128]) + xb,
                           (const char*)&Mp[(size_t)(kk + 32 + k) * N + j0] + xb);
            }
            if (kk + 64 < N) {  // warm GL2 one tile further (global_prefetch_b8)
                __builtin_prefetch(&W2[(size_t)(kk + 64 + (tid >> 3)) * N + i0 + ((tid & 7) << 4)], 0, 1);
                __builtin_prefetch(&Mp[(size_t)(kk + 64 + (tid >> 3)) * N + j0 + ((tid & 7) << 4)], 0, 1);
            }
        }

        // ---- build WMMA fragments: pure ds_load_b128, no per-element VALU
        const bf16_t* A = bfa[cur];
        const bf16_t* B = bfb[cur];
        v16bf afr[2], bfr[4];
#pragma unroll
        for (int mt = 0; mt < 2; ++mt) {
            const int m = waveM * 32 + mt * 16 + lm;
            const v8bf klo = *(const v8bf*)&A[m * LDSB + hi * 8];        // K = 8*hi + 0..7
            const v8bf khi = *(const v8bf*)&A[m * LDSB + 16 + hi * 8];   // K = 16 + 8*hi + 0..7
            afr[mt] = __builtin_shufflevector(klo, khi,
                0, 1, 2, 3, 4, 5, 6, 7, 8, 9, 10, 11, 12, 13, 14, 15);
        }
#pragma unroll
        for (int nt = 0; nt < 4; ++nt) {
            const int n = waveN * 64 + nt * 16 + lm;
            const v8bf klo = *(const v8bf*)&B[n * LDSB + hi * 8];        // K = 8*hi + 0..7
            const v8bf khi = *(const v8bf*)&B[n * LDSB + 16 + hi * 8];   // K = 16 + 8*hi + 0..7
            bfr[nt] = __builtin_shufflevector(klo, khi,
                0, 1, 2, 3, 4, 5, 6, 7, 8, 9, 10, 11, 12, 13, 14, 15);
        }

#pragma unroll
        for (int mt = 0; mt < 2; ++mt)
#pragma unroll
            for (int nt = 0; nt < 4; ++nt)
                acc[mt][nt] = __builtin_amdgcn_wmma_f32_16x16x32_bf16(
                    false, afr[mt], false, bfr[nt], (short)0, acc[mt][nt],
                    false, false);

        // ---- drain async DMA, repack next tile to bf16 (v_cvt_pk_bf16_f32)
        if (more) {
            async_wait0();
            __syncthreads();                    // all waves' DMA landed in stg[nx]
#pragma unroll
            for (int c = 0; c < 8; ++c) {
                const int p  = tid + c * 256;
                const int k2 = (p >> 7) << 1;
                const int x  = p & 127;
                const v2bf pa = {(bf16_t)stg_a[nx][k2 * 128 + x], (bf16_t)stg_a[nx][(k2 + 1) * 128 + x]};
                const v2bf pb = {(bf16_t)stg_b[nx][k2 * 128 + x], (bf16_t)stg_b[nx][(k2 + 1) * 128 + x]};
                *(v2bf*)&bfa[nx][x * LDSB + k2] = pa;
                *(v2bf*)&bfb[nx][x * LDSB + k2] = pb;
            }
        }
        __syncthreads();
        cur ^= 1;
    }

    // ---- write back (C/D layout: VGPR v -> M = v + 8*hi, N = lane%16)
#pragma unroll
    for (int mt = 0; mt < 2; ++mt)
#pragma unroll
        for (int nt = 0; nt < 4; ++nt) {
            const int col = j0 + waveN * 64 + nt * 16 + lm;
#pragma unroll
            for (int v = 0; v < 8; ++v) {
                const int row = i0 + waveM * 32 + mt * 16 + v + hi * 8;
                C[(size_t)row * N + col] = acc[mt][nt][v];
            }
        }
}

// ---------------------------------------------------------------------------
// y[j] += sum_{i in chunk} x[i] * M[i][j]   (row-major M, coalesced over j)
// grid: (m/256, n/chunk); y must be zeroed first.
// ---------------------------------------------------------------------------
__global__ __launch_bounds__(256)
void k_vecmat(float* __restrict__ y, const float* __restrict__ x,
              const float* __restrict__ M, int m, int chunk) {
    const int j    = blockIdx.x * 256 + threadIdx.x;
    const int ibeg = blockIdx.y * chunk;
    float acc = 0.f;
    for (int i = ibeg; i < ibeg + chunk; ++i)
        acc = fmaf(x[i], M[(size_t)i * m + j], acc);
    atomicAdd(&y[j], acc);
}

// ---------------------------------------------------------------------------
// y[r] = dot(W[r][:], x)   one wave32 per row, float4 loads, shuffle reduce
// ---------------------------------------------------------------------------
__global__ __launch_bounds__(256)
void k_matvec(float* __restrict__ y, const float* __restrict__ W,
              const float* __restrict__ x, int rows, int cols) {
    const int gwave = (blockIdx.x * 256 + threadIdx.x) >> 5;
    const int lane  = threadIdx.x & 31;
    if (gwave >= rows) return;
    const float* wr = W + (size_t)gwave * cols;
    float acc = 0.f;
    for (int c = lane * 4; c < cols; c += 128) {
        const float4 wv = *(const float4*)(wr + c);
        const float4 xv = *(const float4*)(x + c);
        acc = fmaf(wv.x, xv.x, acc);
        acc = fmaf(wv.y, xv.y, acc);
        acc = fmaf(wv.z, xv.z, acc);
        acc = fmaf(wv.w, xv.w, acc);
    }
#pragma unroll
    for (int s = 16; s > 0; s >>= 1) acc += __shfl_down(acc, s, 32);
    if (lane == 0) y[gwave] = acc;
}

// ---------------------------------------------------------------------------
// in-place softmax over n elements, single block of 1024 threads
// ---------------------------------------------------------------------------
__global__ __launch_bounds__(1024)
void k_softmax(float* __restrict__ v, int n) {
    __shared__ float red[1024];
    const int tid = threadIdx.x;
    float mx = -INFINITY;
    for (int i = tid; i < n; i += 1024) mx = fmaxf(mx, v[i]);
    red[tid] = mx; __syncthreads();
    for (int s = 512; s > 0; s >>= 1) {
        if (tid < s) red[tid] = fmaxf(red[tid], red[tid + s]);
        __syncthreads();
    }
    mx = red[0]; __syncthreads();
    float sm = 0.f;
    for (int i = tid; i < n; i += 1024) { float e = __expf(v[i] - mx); v[i] = e; sm += e; }
    red[tid] = sm; __syncthreads();
    for (int s = 512; s > 0; s >>= 1) {
        if (tid < s) red[tid] += red[tid + s];
        __syncthreads();
    }
    const float inv = 1.f / red[0];
    for (int i = tid; i < n; i += 1024) v[i] *= inv;
}

// ---------------------------------------------------------------------------
// GRU cell pointwise: h = (1-z)*tanh(i_n + r*h_n) + z*h
// ---------------------------------------------------------------------------
__global__ __launch_bounds__(256)
void k_gru(float* __restrict__ h, const float* __restrict__ gi,
           const float* __restrict__ gh, int n) {
    const int j = blockIdx.x * 256 + threadIdx.x;
    if (j >= n) return;
    const float r  = 1.f / (1.f + __expf(-(gi[j] + gh[j])));
    const float z  = 1.f / (1.f + __expf(-(gi[n + j] + gh[n + j])));
    const float nn = tanhf(gi[2 * n + j] + r * gh[2 * n + j]);
    h[j] = (1.f - z) * nn + z * h[j];
}

// ---------------------------------------------------------------------------
// P_r += softmax3(W3^T @ [sk, xk, |sk-xk|, sk*xk])
// ---------------------------------------------------------------------------
__global__ __launch_bounds__(256)
void k_label(float* __restrict__ Pr, const float* __restrict__ W3,
             const float* __restrict__ sk, const float* __restrict__ xk, int nH) {
    __shared__ float red[3][256];
    const int tid = threadIdx.x;
    float a0 = 0.f, a1 = 0.f, a2 = 0.f;
    for (int f = tid; f < 4 * nH; f += 256) {
        const int seg = f / nH, idx = f % nH;
        float v;
        if      (seg == 0) v = sk[idx];
        else if (seg == 1) v = xk[idx];
        else if (seg == 2) v = fabsf(sk[idx] - xk[idx]);
        else               v = sk[idx] * xk[idx];
        a0 = fmaf(W3[(size_t)f * 3 + 0], v, a0);
        a1 = fmaf(W3[(size_t)f * 3 + 1], v, a1);
        a2 = fmaf(W3[(size_t)f * 3 + 2], v, a2);
    }
    red[0][tid] = a0; red[1][tid] = a1; red[2][tid] = a2; __syncthreads();
    for (int s = 128; s > 0; s >>= 1) {
        if (tid < s) {
            red[0][tid] += red[0][tid + s];
            red[1][tid] += red[1][tid + s];
            red[2][tid] += red[2][tid + s];
        }
        __syncthreads();
    }
    if (tid == 0) {
        const float l0 = red[0][0], l1 = red[1][0], l2 = red[2][0];
        const float mx = fmaxf(l0, fmaxf(l1, l2));
        const float e0 = __expf(l0 - mx), e1 = __expf(l1 - mx), e2 = __expf(l2 - mx);
        const float inv = 1.f / (e0 + e1 + e2);
        Pr[0] += e0 * inv; Pr[1] += e1 * inv; Pr[2] += e2 * inv;
    }
}

__global__ void k_finish(float* __restrict__ out, const float* __restrict__ Pr,
                         const int* __restrict__ Kp) {
    const int l = threadIdx.x;
    if (l < 3) out[l] = Pr[l] / (float)(*Kp);
}

// ---------------------------------------------------------------------------
// Host orchestration
// ---------------------------------------------------------------------------
extern "C" void kernel_launch(void* const* d_in, const int* in_sizes, int n_in,
                              void* d_out, int out_size, void* d_ws, size_t ws_size,
                              hipStream_t stream) {
    (void)in_sizes; (void)n_in; (void)out_size; (void)ws_size;
    const float* M_h  = (const float*)d_in[0];
    const float* M_p  = (const float*)d_in[1];
    const float* w1   = (const float*)d_in[2];
    const float* W2   = (const float*)d_in[3];
    const float* W3   = (const float*)d_in[4];
    const float* w_ih = (const float*)d_in[5];
    const float* w_hh = (const float*)d_in[6];
    const int*   Kp   = (const int*)d_in[7];
    float* out = (float*)d_out;

    // workspace layout
    float* C  = (float*)d_ws;                       // 4096*4096 f32 (64 MB)
    float* t  = C  + (size_t)NHID * NTP;            // logits / softmax buffer
    float* sk = t  + NTH;                           // hidden state (4096)
    float* xk = sk + NTH;                           // GRU input   (4096)
    float* gi = xk + NTP;                           // 3*4096
    float* gh = gi + 3 * NTH;                       // 3*4096
    float* Pr = gh + 3 * NTH;                       // 3 (+1 pad)

    const dim3 blk256(256), blk1024(1024);
    const dim3 gVecmat(NTH / 256, 32);              // split-K over 32 chunks of 128 rows
    const dim3 gGemm(NTP / 128, NHID / 128);        // 32 x 32 tiles of 128x128
    const dim3 gMatvec((3 * NTH) / 8);              // one wave per row, 8 rows/block
    const dim3 gGru(NTH / 256);

    // alpha = softmax(w1^T M_h); sk0 = alpha @ M_h
    hipMemsetAsync(t, 0, NTH * sizeof(float), stream);
    k_vecmat<<<gVecmat, blk256, 0, stream>>>(t, w1, M_h, NTH, 128);
    k_softmax<<<1, blk1024, 0, stream>>>(t, NTH);
    hipMemsetAsync(sk, 0, NTH * sizeof(float), stream);
    k_vecmat<<<gVecmat, blk256, 0, stream>>>(sk, t, M_h, NTH, 128);

    // C = W2^T @ M_p  (the WMMA GEMM)
    k_wmma_gemm<<<gGemm, blk256, 0, stream>>>(C, W2, M_p, NHID);

    hipMemsetAsync(Pr, 0, 4 * sizeof(float), stream);

    // K_STEPS = 8 (reference module constant; host loop count must be static)
    for (int step = 0; step < 8; ++step) {
        hipMemsetAsync(t, 0, NTP * sizeof(float), stream);
        k_vecmat<<<gVecmat, blk256, 0, stream>>>(t, sk, C, NTP, 128);   // sk @ C
        k_softmax<<<1, blk1024, 0, stream>>>(t, NTP);                   // beta
        hipMemsetAsync(xk, 0, NTP * sizeof(float), stream);
        k_vecmat<<<gVecmat, blk256, 0, stream>>>(xk, t, M_p, NTP, 128); // beta @ M_p
        k_matvec<<<gMatvec, blk256, 0, stream>>>(gi, w_ih, xk, 3 * NTH, NTP);
        k_matvec<<<gMatvec, blk256, 0, stream>>>(gh, w_hh, sk, 3 * NTH, NTH);
        k_gru<<<gGru, blk256, 0, stream>>>(sk, gi, gh, NTH);            // sk <- h_new
        k_label<<<1, blk256, 0, stream>>>(Pr, W3, sk, xk, NTH);
    }
    k_finish<<<1, 32, 0, stream>>>(out, Pr, Kp);
}
